// outcome_detection_model_39462159515657
// MI455X (gfx1250) — matrix operand order repaired and checked by reading, not compiled, not run on hardware
//
#include <hip/hip_runtime.h>
#include <hip/hip_bf16.h>

#define H 1024
#define D 200
#define K 8
#define B 8
#define L 4096
#define NPAD 208        // 200 cols + mean col (200) + 7 zero pad
#define NT 13           // 13 n-tiles of 16
#define KC 32           // K chunk per WMMA pass
#define MT 128          // token rows per workgroup
#define XSTRIDE 34      // padded LDS halfword stride (conflict-free; == TDM pad 16dw+1dw)

typedef __attribute__((ext_vector_type(16))) __bf16 v16bf;
typedef __attribute__((ext_vector_type(8)))  float  v8f;
typedef __attribute__((ext_vector_type(4))) unsigned v4u;
typedef __attribute__((ext_vector_type(8))) int      v8i;
typedef __attribute__((ext_vector_type(4))) int      v4i;

union FragBF { v16bf v; unsigned u[8]; };

__device__ __forceinline__ unsigned short f32_to_bf16(float f) {
  union { float f; unsigned u; } c; c.f = f;
  unsigned u = c.u;
  return (unsigned short)((u + 0x7FFFu + ((u >> 16) & 1u)) >> 16);
}

__device__ __forceinline__ void tdm_load_2d(v4u g0, v8i g1) {
  v4i gz4 = {0, 0, 0, 0};
#if __clang_major__ >= 23
  v8i gz8 = {0, 0, 0, 0, 0, 0, 0, 0};
  __builtin_amdgcn_tensor_load_to_lds(g0, g1, gz4, gz4, gz8, 0);
#else
  __builtin_amdgcn_tensor_load_to_lds(g0, g1, gz4, gz4, 0);
#endif
}

// -------- Kernel 0: build W1^T padded bf16 [NPAD][H]; row 200 = 1/H, rows 201..207 = 0
__global__ __launch_bounds__(256) void prep_w1_kernel(
    const float* __restrict__ w1, unsigned short* __restrict__ w1t) {
  int idx = blockIdx.x * 256 + threadIdx.x;      // over H*NPAD, n fastest (coalesced w1 reads)
  if (idx >= H * NPAD) return;
  int h = idx / NPAD, n = idx - h * NPAD;
  float v = (n < D) ? w1[h * D + n] : ((n == D) ? (1.0f / (float)H) : 0.0f);
  w1t[(size_t)n * H + h] = f32_to_bf16(v);
}

// -------- Kernel 1: fused GEMM1(tanh) + GEMM2 + mean-scale -> s ---------------
__global__ __launch_bounds__(256) void gemm_s_kernel(
    const float* __restrict__ x,              // [B*L, H] f32
    const unsigned short* __restrict__ w1t,   // [NPAD, H] bf16 (transposed, padded)
    const float* __restrict__ b1,             // [D]
    const float* __restrict__ w2,             // [D, K]
    const float* __restrict__ b2,             // [K]
    float* __restrict__ sbuf)                 // [B*L, K] f32
{
  __shared__ __align__(16) unsigned short lds_x[MT * XSTRIDE];     //  8704 B
  __shared__ __align__(16) unsigned short lds_w[NPAD * XSTRIDE];   // 14144 B (TDM-padded rows)
  __shared__ float lds_b1[NPAD];
  __shared__ float lds_w2[NPAD * K];
  __shared__ float lds_b2[K];

  const int tid  = threadIdx.x;
  const int lane = tid & 31;
  const int wave = tid >> 5;
  const int m    = lane & 15;
  const int half = lane >> 4;
  const int tileBase = blockIdx.x * MT;

  // stage padded b1 / W2 / b2 once
  for (int i = tid; i < NPAD; i += 256) lds_b1[i] = (i < D) ? b1[i] : 0.0f;
  for (int i = tid; i < NPAD * K; i += 256) {
    int d = i >> 3, k = i & 7;
    lds_w2[i] = (d < D) ? w2[d * K + k] : 0.0f;
  }
  if (tid < K) lds_b2[tid] = b2[tid];

  const unsigned lds_w_base = (unsigned)(size_t)&lds_w[0];
  const unsigned long long w1t_base = (unsigned long long)(size_t)w1t;

  v8f acc[NT];
  #pragma unroll
  for (int nt = 0; nt < NT; ++nt)
    #pragma unroll
    for (int i = 0; i < 8; ++i) acc[nt][i] = 0.0f;

  for (int kc = 0; kc < H; kc += KC) {
    __syncthreads();

    // ---- TDM: DMA the W1^T tile (NPAD rows x 32 bf16) into lds_w, one wave issues.
    // D# per ISA ch.8: data_size=2B, tile 32 x 208, tensor stride0 = H elements,
    // LDS pad: 1 DWORD after every 16 DWORDs -> 68B row stride == XSTRIDE.
    if (tid < 32) {
      unsigned long long ga = w1t_base + (unsigned)(kc * 2);
      v4u g0;
      g0.x = 1u;                                            // count=1, user mode
      g0.y = lds_w_base;                                    // lds_addr (bytes)
      g0.z = (unsigned)(ga & 0xFFFFFFFFull);                // global_addr[31:0]
      g0.w = (unsigned)((ga >> 32) & 0x01FFFFFFull) | (2u << 30);  // addr[56:32] | type=2
      v8i g1;
      g1[0] = (1 << 16) | (1 << 20) | (3 << 22);            // data_size=2B, pad_en, ivl=16dw, amt=1dw
      g1[1] = (int)(H << 16);                               // tensor_dim0[15:0]=1024
      g1[2] = (int)(NPAD << 16);                            // dim0 hi=0 | tensor_dim1=208
      g1[3] = (int)(KC << 16);                              // dim1 hi=0 | tile_dim0=32
      g1[4] = NPAD;                                         // tile_dim1=208, tile_dim2=0
      g1[5] = H;                                            // tensor_dim0_stride=1024
      g1[6] = 0;
      g1[7] = 0;
      tdm_load_2d(g0, g1);
    }

    // ---- stage X chunk: MT x KC f32 -> bf16 (coalesced reads, VALU convert)
    #pragma unroll
    for (int i = 0; i < (MT * KC) / 256; ++i) {
      int idx = tid + 256 * i;
      int r = idx >> 5, k = idx & 31;
      const float* gp = &x[(size_t)(tileBase + r) * H + kc + k];
      lds_x[r * XSTRIDE + k] = f32_to_bf16(*gp);
      if (kc + KC < H) __builtin_prefetch(gp + KC, 0, 1);  // global_prefetch_b8
    }
    if (tid < 32) __builtin_amdgcn_s_wait_tensorcnt(0);    // TDM writes visible
    __syncthreads();

    // A fragment: 16x32 bf16, rows = wave's 16 tokens (ISA 7.12.2 layout)
    FragBF a;
    {
      const unsigned* lx = (const unsigned*)lds_x;
      int row = (wave << 4) + m;
      int base = half << 3;  // 0 or 8
      #pragma unroll
      for (int p = 0; p < 8; ++p) {
        int k0 = base + ((p < 4) ? (p << 1) : (16 + ((p - 4) << 1)));
        a.u[p] = lx[(row * XSTRIDE + k0) >> 1];
      }
    }
    const unsigned* lw = (const unsigned*)lds_w;
    #pragma unroll
    for (int nt = 0; nt < NT; ++nt) {
      FragBF bf;
      int n = (nt << 4) + m;
      #pragma unroll
      for (int p = 0; p < 8; ++p) {
        int k0 = (half << 3) + ((p < 4) ? (p << 1) : (16 + ((p - 4) << 1)));       // B: lanes0-15 K=0..7,16..23, lanes16-31 K=8..15,24..31
        bf.u[p] = lw[(n * XSTRIDE + k0) >> 1];
      }
      acc[nt] = __builtin_amdgcn_wmma_f32_16x16x32_bf16(
          false, a.v, false, bf.v, (short)0, acc[nt], false, false);
    }
  }
  __syncthreads();

  // Epilogue: C/D layout -> VGPR r holds M = r + 8*half, N = 16*nt + m.
  // attn = tanh(acc + b1); s_k = (sum_n attn*W2 + b2[k]) * mean ; mean = col 200 (nt=12, m==8)
  #pragma unroll
  for (int r = 0; r < 8; ++r) {
    float mrow = __shfl(acc[12][r], (lane & 16) + 8, 32);
    float attn[NT];
    #pragma unroll
    for (int nt = 0; nt < NT; ++nt) {
      int n = (nt << 4) + m;
      attn[nt] = tanhf(acc[nt][r] + lds_b1[n]);
    }
    int rowg = tileBase + (wave << 4) + (half << 3) + r;
    #pragma unroll
    for (int k = 0; k < K; ++k) {
      float local = 0.0f;
      #pragma unroll
      for (int nt = 0; nt < NT; ++nt) {
        int n = (nt << 4) + m;
        local += attn[nt] * lds_w2[(n << 3) + k];
      }
      local += __shfl_xor(local, 1, 32);
      local += __shfl_xor(local, 2, 32);
      local += __shfl_xor(local, 4, 32);
      local += __shfl_xor(local, 8, 32);
      float sv = (local + lds_b2[k]) * mrow;
      if (m == k) sbuf[(size_t)rowg * K + k] = sv;
    }
  }
}

// -------- Kernel 2: per-batch scan + det-select + softmax loss ----------------
__global__ __launch_bounds__(256) void scan_loss_kernel(
    const float* __restrict__ sbuf, const int* __restrict__ prompt_types,
    const int* __restrict__ labels, float* __restrict__ out)
{
  __shared__ float lds[256 * K];
  __shared__ float lred[256];
  const int b = blockIdx.x;
  const int tid = threadIdx.x;
  const int pt = prompt_types[b];
  const float* sb = sbuf + (size_t)b * L * K;
  const int l0 = tid * (L / 256);     // 16 positions per thread

  float loc[K];
  #pragma unroll
  for (int k = 0; k < K; ++k) loc[k] = 0.0f;
  for (int i = 0; i < L / 256; ++i) {
    const float* p = sb + (size_t)(l0 + i) * K;
    #pragma unroll
    for (int k = 0; k < K; ++k) loc[k] += p[k];
  }
  #pragma unroll
  for (int k = 0; k < K; ++k) lds[tid * K + k] = loc[k];
  __syncthreads();

  for (int off = 1; off < 256; off <<= 1) {
    float t[K];
    if (tid >= off) {
      #pragma unroll
      for (int k = 0; k < K; ++k) t[k] = lds[(tid - off) * K + k];
    }
    __syncthreads();
    if (tid >= off) {
      #pragma unroll
      for (int k = 0; k < K; ++k) lds[tid * K + k] += t[k];
    }
    __syncthreads();
  }

  float run[K], total[K];
  #pragma unroll
  for (int k = 0; k < K; ++k) {
    run[k]   = lds[tid * K + k] - loc[k];   // exclusive prefix for this chunk
    total[k] = lds[255 * K + k];
  }
  __syncthreads();

  float lacc = 0.0f;
  for (int i = 0; i < L / 256; ++i) {
    int l = l0 + i;
    const float* p = sb + (size_t)l * K;
    float inv_pre  = 1.0f / (float)(l + 1);
    float inv_post = 1.0f / (float)(L - l);
    float det[K];
    #pragma unroll
    for (int k = 0; k < K; ++k) {
      float sval = p[k];
      float runb = run[k];
      run[k] = runb + sval;
      float pre  = run[k] * inv_pre;
      float post = (total[k] - runb) * inv_post;
      det[k] = (pt == 0) ? pre : ((pt == 1) ? post : sval);
    }
    float mx = det[0];
    #pragma unroll
    for (int k = 1; k < K; ++k) mx = fmaxf(mx, det[k]);
    float e[K], sum = 0.0f;
    #pragma unroll
    for (int k = 0; k < K; ++k) { e[k] = expf(det[k] - mx); sum += e[k]; }
    float inv = 1.0f / sum;
    float pk[K];
    #pragma unroll
    for (int k = 0; k < K; ++k) pk[k] = e[k] * inv;
    float mx2 = pk[0];
    #pragma unroll
    for (int k = 1; k < K; ++k) mx2 = fmaxf(mx2, pk[k]);
    float s2 = 0.0f;
    #pragma unroll
    for (int k = 0; k < K; ++k) s2 += expf(pk[k] - mx2);
    float lse = mx2 + logf(s2);
    int lab = labels[b * L + l] & 7;
    lacc += pk[lab] - lse;
  }
  lred[tid] = lacc;
  __syncthreads();
  for (int off = 128; off > 0; off >>= 1) {
    if (tid < off) lred[tid] += lred[tid + off];
    __syncthreads();
  }
  if (tid == 0) out[b] = -lred[0] / (float)L;
}

extern "C" void kernel_launch(void* const* d_in, const int* in_sizes, int n_in,
                              void* d_out, int out_size, void* d_ws, size_t ws_size,
                              hipStream_t stream) {
  const float* hidd = (const float*)d_in[0];   // (B,L,H) f32
  const float* W1   = (const float*)d_in[1];   // (H,D)
  const float* b1   = (const float*)d_in[2];   // (D,)
  const float* W2   = (const float*)d_in[3];   // (D,K)
  const float* b2   = (const float*)d_in[4];   // (K,)
  const int*   pt   = (const int*)d_in[5];     // (B,)
  const int*   lab  = (const int*)d_in[6];     // (B,L)
  float* out = (float*)d_out;                  // (B,)

  unsigned short* w1t = (unsigned short*)d_ws;                 // NPAD*H bf16 = 416 KB
  float* sbuf = (float*)((char*)d_ws + (size_t)(1 << 19));     // B*L*K f32 = 1 MB @ 512 KB

  prep_w1_kernel<<<(H * NPAD + 255) / 256, 256, 0, stream>>>(W1, w1t);
  gemm_s_kernel<<<(B * L) / MT, 256, 0, stream>>>(hidd, w1t, b1, W2, b2, sbuf);
  scan_loss_kernel<<<B, 256, 0, stream>>>(sbuf, pt, lab, out);
}
